// Dictionnary_35777077575854
// MI455X (gfx1250) — matrix operand order repaired and checked by reading, not compiled, not run on hardware
//
#include <hip/hip_runtime.h>

// ---------------------------------------------------------------------------
// Problem constants (from the reference)
// ---------------------------------------------------------------------------
#define NB_ATOMS 128
#define ATOM     12
#define AA       144          // ATOM*ATOM
#define KPAD     160          // AA padded to multiple of 32 for WMMA K
#define BATCH    8
#define HW       75
#define PG       64           // patch grid (HW-ATOM+1)
#define NPATCH   32768        // BATCH*PG*PG
#define LAM      0.1f
#define FISTA_ITERS 15

typedef __attribute__((ext_vector_type(16))) __bf16 v16bf;
typedef __attribute__((ext_vector_type(8)))  float  v8f;

union BFrag { uint4 q[2]; v16bf v; };

static __device__ __forceinline__ unsigned short f2bf(float x) {
    unsigned int u = __float_as_uint(x);
    unsigned int r = (u + 0x7FFFu + ((u >> 16) & 1u)) >> 16;
    return (unsigned short)r;
}

static __device__ __forceinline__ v8f wmma_bf(const BFrag& a, const BFrag& b, v8f c) {
    return __builtin_amdgcn_wmma_f32_16x16x32_bf16(false, a.v, false, b.v,
                                                   (short)0, c, false, false);
}

// ---------------------------------------------------------------------------
// K0: normalize atoms, gram, spectral norm (power iteration), emit bf16 views
//   gX    [128][128]  bf16  : X = gram / (lam*mu)
//   gAfn  [128][160]  bf16  : normalized atoms, K-padded with zeros
//   gAfnT [160][128]  bf16  : transpose of the above
// ---------------------------------------------------------------------------
__global__ __launch_bounds__(256)
void prep_atoms(const float* __restrict__ atoms, const float* __restrict__ mu_p,
                unsigned short* __restrict__ gX,
                unsigned short* __restrict__ gAfn,
                unsigned short* __restrict__ gAfnT) {
    extern __shared__ unsigned char smem[];
    float* Af   = (float*)smem;                       // 128*144 = 73728 B
    float* X0   = (float*)(smem + 73728);             // 128*128 = 65536 B
    float* vv   = (float*)(smem + 73728 + 65536);     // 128
    float* ww   = vv + 128;                           // 128
    float* scal = ww + 128;                           // 1

    const int tid = threadIdx.x;
    for (int i = tid; i < NB_ATOMS * AA; i += 256) Af[i] = atoms[i];
    __syncthreads();

    if (tid < NB_ATOMS) {
        float s = 0.f;
        for (int k = 0; k < AA; ++k) s += Af[tid * AA + k];
        float mean = s * (1.0f / AA);
        float ss = 0.f;
        for (int k = 0; k < AA; ++k) {
            float d = Af[tid * AA + k] - mean;
            Af[tid * AA + k] = d; ss += d * d;
        }
        float inv = rsqrtf(fmaxf(ss, 1e-30f));
        for (int k = 0; k < AA; ++k) Af[tid * AA + k] *= inv;
    }
    __syncthreads();

    // gram X0 = Af @ Af^T
    for (int i = tid; i < NB_ATOMS * NB_ATOMS; i += 256) {
        int r = i >> 7, c = i & 127;
        float s = 0.f;
        for (int k = 0; k < AA; ++k) s += Af[r * AA + k] * Af[c * AA + k];
        X0[i] = s;
    }
    if (tid < NB_ATOMS) vv[tid] = 1.0f;
    __syncthreads();

    // power iteration for lam = lambda_max(X0) = sigma^2
    for (int itp = 0; itp < 160; ++itp) {
        if (tid < NB_ATOMS) {
            float s = 0.f;
            for (int k = 0; k < NB_ATOMS; ++k) s += X0[tid * 128 + k] * vv[k];
            ww[tid] = s;
        }
        __syncthreads();
        if (tid == 0) {
            float s = 0.f;
            for (int k = 0; k < NB_ATOMS; ++k) s += ww[k] * ww[k];
            scal[0] = sqrtf(s);
        }
        __syncthreads();
        if (tid < NB_ATOMS) vv[tid] = ww[tid] / fmaxf(scal[0], 1e-30f);
        __syncthreads();
    }

    const float lam = fmaxf(scal[0], 1e-30f);
    const float mu  = fmaxf(mu_p[0], 0.0f);
    const float sx  = 1.0f / (lam * mu);
    const float sa  = rsqrtf(lam * mu);

    for (int i = tid; i < NB_ATOMS * NB_ATOMS; i += 256) gX[i] = f2bf(X0[i] * sx);
    for (int i = tid; i < NB_ATOMS * KPAD; i += 256) {
        int n = i / KPAD, k = i - n * KPAD;
        float v = (k < AA) ? Af[n * AA + k] * sa : 0.0f;
        unsigned short bv = f2bf(v);
        gAfn[i] = bv;
        gAfnT[k * NB_ATOMS + n] = bv;
    }
}

// ---------------------------------------------------------------------------
// K1: per-patch mean of y (computed once, reused both unroll iterations)
// ---------------------------------------------------------------------------
__global__ __launch_bounds__(256)
void patch_mean_k(const float* __restrict__ y, float* __restrict__ pm) {
    int p = blockIdx.x * 256 + threadIdx.x;
    if (p >= NPATCH) return;
    int b = p >> 12, r = p & 4095;
    int ph = r >> 6, pw = r & 63;
    const float* base = y + b * (HW * HW) + ph * HW + pw;
    float s = 0.f;
    for (int ki = 0; ki < ATOM; ++ki)
        for (int kj = 0; kj < ATOM; ++kj)
            s += base[ki * HW + kj];
    pm[p] = s * (1.0f / AA);
}

// ---------------------------------------------------------------------------
// K2: fused q-GEMM + full 15-iter FISTA + final prox step + pred-GEMM.
//   One workgroup = 64 patches (one (b, ph) row of the patch grid).
//   8 waves; wave w owns Mtile mt=w/2 and Ntiles (w&1)*4..+3.
//   All WMMA: bf16 inputs, f32 accumulate (v_wmma_f32_16x16x32_bf16).
// ---------------------------------------------------------------------------
__global__ __launch_bounds__(256)
void fused_fista(const float* __restrict__ goal,
                 const float* __restrict__ pm,
                 const unsigned short* __restrict__ gAfn,
                 const unsigned short* __restrict__ gAfnT,
                 const unsigned short* __restrict__ gX,
                 float* __restrict__ c_buf,
                 float* __restrict__ pred,
                 const float* __restrict__ mu_p,
                 int init_c) {
    extern __shared__ unsigned char smem[];
    unsigned short* sAfn  = (unsigned short*)(smem);            // 128*160 = 40960 B
    unsigned short* sAfnT = (unsigned short*)(smem + 40960);    // 160*128 = 40960 B
    unsigned short* sX    = (unsigned short*)(smem + 81920);    // 128*128 = 32768 B
    unsigned short* sZ    = (unsigned short*)(smem + 114688);   //  64*128 = 16384 B
    unsigned short* sP    = (unsigned short*)(smem + 131072);   //  64*160 = 20480 B
    float*          sGoal = (float*)(smem + 151552);            //  12*80  =  3840 B

    const int tid  = threadIdx.x;
    const int wg   = blockIdx.x;            // 0..511 = b*64 + ph
    const int b    = wg >> 6;
    const int ph   = wg & 63;
    const int pbase = wg * 64;
    const int lane = tid & 31;
    const int wv   = tid >> 5;
    const int h    = lane >> 4;             // lane half (wave32)
    const int ln   = lane & 15;
    const float mu = fmaxf(mu_p[0], 0.0f);

    // ---- stage dictionary / gram / goal rows into LDS ----
    {
        const unsigned int* g0 = (const unsigned int*)gAfn;
        unsigned int* s0 = (unsigned int*)sAfn;
        for (int i = tid; i < NB_ATOMS * KPAD / 2; i += 256) s0[i] = g0[i];
        const unsigned int* g1 = (const unsigned int*)gAfnT;
        unsigned int* s1 = (unsigned int*)sAfnT;
        for (int i = tid; i < KPAD * NB_ATOMS / 2; i += 256) s1[i] = g1[i];
        const unsigned int* g2 = (const unsigned int*)gX;
        unsigned int* s2 = (unsigned int*)sX;
        for (int i = tid; i < NB_ATOMS * NB_ATOMS / 2; i += 256) s2[i] = g2[i];
    }
    for (int i = tid; i < ATOM * HW; i += 256) {
        int r = i / HW, c = i - r * HW;
        sGoal[r * 80 + c] = goal[b * (HW * HW) + (ph + r) * HW + c];
    }
    __syncthreads();

    // ---- im2col into bf16 P tile [64 patches][160 Kpad] ----
    for (int i = tid; i < 64 * KPAD; i += 256) {
        int m = i / KPAD, k = i - m * KPAD;
        float v = 0.0f;
        if (k < AA) { int ki = k / ATOM, kj = k - ki * ATOM; v = sGoal[ki * 80 + m + kj]; }
        sP[i] = f2bf(v);
    }
    __syncthreads();

    const int mt  = wv >> 1;                // 0..3  (16-row M tile)
    const int ntg = (wv & 1) * 4;           // N tile group base

    v8f zf = {};
    v8f qT[4]; v8f cT[4]; v8f zT[4];

    // ---- q = im2col(goal) @ Af^T,  K = 160 (last 16 are zero pad) ----
    qT[0] = zf; qT[1] = zf; qT[2] = zf; qT[3] = zf;
    #pragma unroll
    for (int kk = 0; kk < 5; ++kk) {
        BFrag A;
        const unsigned short* ap = &sP[(mt * 16 + ln) * KPAD + kk * 32 + h * 16];
        A.q[0] = *(const uint4*)ap;
        A.q[1] = *(const uint4*)(ap + 8);
        #pragma unroll
        for (int j = 0; j < 4; ++j) {
            BFrag B;
            const unsigned short* bp = &sAfn[((ntg + j) * 16 + ln) * KPAD + kk * 32 + h * 16];
            B.q[0] = *(const uint4*)bp;
            B.q[1] = *(const uint4*)(bp + 8);
            qT[j] = wmma_bf(A, B, qT[j]);
        }
    }

    // ---- load / init c, z (C-fragment layout: elem e -> row e+8h, col ln) ----
    if (init_c) {
        cT[0] = zf; cT[1] = zf; cT[2] = zf; cT[3] = zf;
    } else {
        #pragma unroll
        for (int j = 0; j < 4; ++j)
            #pragma unroll
            for (int e = 0; e < 8; ++e) {
                int p = pbase + mt * 16 + e + 8 * h;
                int n = (ntg + j) * 16 + ln;
                cT[j][e] = c_buf[p * NB_ATOMS + n];
            }
    }
    zT[0] = cT[0]; zT[1] = cT[1]; zT[2] = cT[2]; zT[3] = cT[3];

    // ---- 15 FISTA iterations, entirely in-kernel ----
    float t = 1.0f;
    #pragma unroll 1
    for (int it = 0; it < FISTA_ITERS; ++it) {
        __syncthreads();
        #pragma unroll
        for (int j = 0; j < 4; ++j)
            #pragma unroll
            for (int e = 0; e < 8; ++e)
                sZ[(mt * 16 + e + 8 * h) * NB_ATOMS + (ntg + j) * 16 + ln] = f2bf(zT[j][e]);
        __syncthreads();

        v8f uT[4]; uT[0] = zf; uT[1] = zf; uT[2] = zf; uT[3] = zf;
        #pragma unroll
        for (int kk = 0; kk < 4; ++kk) {
            BFrag A;
            const unsigned short* ap = &sZ[(mt * 16 + ln) * NB_ATOMS + kk * 32 + h * 16];
            A.q[0] = *(const uint4*)ap;
            A.q[1] = *(const uint4*)(ap + 8);
            #pragma unroll
            for (int j = 0; j < 4; ++j) {
                BFrag B;   // X is symmetric: B[k][n] = X[n][k], contiguous in k
                const unsigned short* bp = &sX[((ntg + j) * 16 + ln) * NB_ATOMS + kk * 32 + h * 16];
                B.q[0] = *(const uint4*)bp;
                B.q[1] = *(const uint4*)(bp + 8);
                uT[j] = wmma_bf(A, B, uT[j]);
            }
        }
        float tn = 0.5f * (1.0f + sqrtf(1.0f + 4.0f * t * t));
        float wm = (t - 1.0f) / tn;
        #pragma unroll
        for (int j = 0; j < 4; ++j)
            #pragma unroll
            for (int e = 0; e < 8; ++e) {
                float g  = zT[j][e] - (uT[j][e] - qT[j][e]) * mu;
                float cn = copysignf(fmaxf(fabsf(g) - LAM, 0.0f), g);
                zT[j][e] = cn + wm * (cn - cT[j][e]);
                cT[j][e] = cn;
            }
        t = tn;
    }

    // ---- final differentiable step: cf = prox(c - (c@X - q)*mu) ----
    __syncthreads();
    #pragma unroll
    for (int j = 0; j < 4; ++j)
        #pragma unroll
        for (int e = 0; e < 8; ++e)
            sZ[(mt * 16 + e + 8 * h) * NB_ATOMS + (ntg + j) * 16 + ln] = f2bf(cT[j][e]);
    __syncthreads();
    {
        v8f uT[4]; uT[0] = zf; uT[1] = zf; uT[2] = zf; uT[3] = zf;
        #pragma unroll
        for (int kk = 0; kk < 4; ++kk) {
            BFrag A;
            const unsigned short* ap = &sZ[(mt * 16 + ln) * NB_ATOMS + kk * 32 + h * 16];
            A.q[0] = *(const uint4*)ap;
            A.q[1] = *(const uint4*)(ap + 8);
            #pragma unroll
            for (int j = 0; j < 4; ++j) {
                BFrag B;
                const unsigned short* bp = &sX[((ntg + j) * 16 + ln) * NB_ATOMS + kk * 32 + h * 16];
                B.q[0] = *(const uint4*)bp;
                B.q[1] = *(const uint4*)(bp + 8);
                uT[j] = wmma_bf(A, B, uT[j]);
            }
        }
        #pragma unroll
        for (int j = 0; j < 4; ++j)
            #pragma unroll
            for (int e = 0; e < 8; ++e) {
                float g = cT[j][e] - (uT[j][e] - qT[j][e]) * mu;
                cT[j][e] = copysignf(fmaxf(fabsf(g) - LAM, 0.0f), g);
            }
    }
    // persist cf for the next unroll iteration
    #pragma unroll
    for (int j = 0; j < 4; ++j)
        #pragma unroll
        for (int e = 0; e < 8; ++e) {
            int p = pbase + mt * 16 + e + 8 * h;
            c_buf[p * NB_ATOMS + (ntg + j) * 16 + ln] = cT[j][e];
        }

    // ---- pred = cf @ atoms_n  ([64,128] x [128,144]) + patch_mean ----
    __syncthreads();
    #pragma unroll
    for (int j = 0; j < 4; ++j)
        #pragma unroll
        for (int e = 0; e < 8; ++e)
            sZ[(mt * 16 + e + 8 * h) * NB_ATOMS + (ntg + j) * 16 + ln] = f2bf(cT[j][e]);
    __syncthreads();

    for (int tix = wv; tix < 36; tix += 8) {     // 4 Mtiles x 9 Ntiles
        int m2 = tix / 9, n2 = tix - m2 * 9;
        v8f acc = zf;
        #pragma unroll
        for (int kk = 0; kk < 4; ++kk) {
            BFrag A;
            const unsigned short* ap = &sZ[(m2 * 16 + ln) * NB_ATOMS + kk * 32 + h * 16];
            A.q[0] = *(const uint4*)ap;
            A.q[1] = *(const uint4*)(ap + 8);
            BFrag B;   // B[k=atom][N=pixel] = AfnT[pixel][atom], contiguous in atom
            const unsigned short* bp = &sAfnT[(n2 * 16 + ln) * NB_ATOMS + kk * 32 + h * 16];
            B.q[0] = *(const uint4*)bp;
            B.q[1] = *(const uint4*)(bp + 8);
            acc = wmma_bf(A, B, acc);
        }
        #pragma unroll
        for (int e = 0; e < 8; ++e) {
            int p = pbase + m2 * 16 + e + 8 * h;
            pred[p * AA + n2 * 16 + ln] = acc[e] + pm[p];
        }
    }
}

// ---------------------------------------------------------------------------
// K3: deterministic gather fold + goal update.
//   out[b,i,j] = (y + beta * sum_patches pred) / (1 + beta * overlap_count)
// ---------------------------------------------------------------------------
__global__ __launch_bounds__(256)
void fold_k(const float* __restrict__ pred, const float* __restrict__ y,
            const float* __restrict__ beta_p, float* __restrict__ out) {
    int t = blockIdx.x * 256 + threadIdx.x;
    if (t >= BATCH * HW * HW) return;
    int b = t / (HW * HW), r = t - b * (HW * HW);
    int i = r / HW, j = r - i * HW;
    int pi0 = max(0, i - (ATOM - 1)), pi1 = min(PG - 1, i);
    int pj0 = max(0, j - (ATOM - 1)), pj1 = min(PG - 1, j);
    float s = 0.f;
    for (int pi = pi0; pi <= pi1; ++pi) {
        const float* row = pred + (((b << 6) + pi) * PG) * AA + (i - pi) * ATOM;
        for (int pj = pj0; pj <= pj1; ++pj)
            s += row[pj * AA + (j - pj)];
    }
    float beta = fmaxf(beta_p[0], 0.0f);
    float div  = (float)((pi1 - pi0 + 1) * (pj1 - pj0 + 1));
    out[t] = (y[t] + beta * s) / (1.0f + beta * div);
}

// ---------------------------------------------------------------------------
// Host launcher
// ---------------------------------------------------------------------------
extern "C" void kernel_launch(void* const* d_in, const int* in_sizes, int n_in,
                              void* d_out, int out_size, void* d_ws, size_t ws_size,
                              hipStream_t stream) {
    const float* y     = (const float*)d_in[0];
    const float* atoms = (const float*)d_in[1];
    const float* beta  = (const float*)d_in[2];
    const float* mu    = (const float*)d_in[3];

    char* ws = (char*)d_ws;
    unsigned short* gX    = (unsigned short*)(ws + 0);        //  32768 B
    unsigned short* gAfn  = (unsigned short*)(ws + 32768);    //  40960 B
    unsigned short* gAfnT = (unsigned short*)(ws + 73728);    //  40960 B
    float* pm    = (float*)(ws + 114688);                     // 131072 B
    float* goal1 = (float*)(ws + 245760);                     // 180000 B
    float* c_buf = (float*)(ws + 425984);                     // 16 MB
    float* pred  = (float*)(ws + 425984 + 16777216);          // ~18 MB
    float* out   = (float*)d_out;

    const size_t lds_prep  = 73728 + 65536 + 2 * 512 + 64;
    const size_t lds_fused = 155392;

    hipLaunchKernelGGL(prep_atoms, dim3(1), dim3(256), lds_prep, stream,
                       atoms, mu, gX, gAfn, gAfnT);
    hipLaunchKernelGGL(patch_mean_k, dim3(128), dim3(256), 0, stream, y, pm);

    // unroll iteration 0: goal = y, c starts at zero
    hipLaunchKernelGGL(fused_fista, dim3(512), dim3(256), lds_fused, stream,
                       y, pm, gAfn, gAfnT, gX, c_buf, pred, mu, 1);
    hipLaunchKernelGGL(fold_k, dim3(176), dim3(256), 0, stream, pred, y, beta, goal1);

    // unroll iteration 1: goal = goal1, c carried over
    hipLaunchKernelGGL(fused_fista, dim3(512), dim3(256), lds_fused, stream,
                       goal1, pm, gAfn, gAfnT, gX, c_buf, pred, mu, 0);
    hipLaunchKernelGGL(fold_k, dim3(176), dim3(256), 0, stream, pred, y, beta, out);
}
